// GatedSpikingReservoirStep_10806137717145
// MI455X (gfx1250) — matrix operand order repaired and checked by reading, not compiled, not run on hardware
//
#include <hip/hip_runtime.h>
#include <hip/hip_bf16.h>
#include <stdint.h>

// ---------------------------------------------------------------------------
// GatedSpikingReservoirStep for MI455X (gfx1250, wave32, WMMA, async-LDS)
//   S    = [inputs | prev] @ [W_in | W_res]^T          (K = 3072)
//   i/f/o= sigmoid(inputs @ W_gate[g]^T)               (K = 1024)
//   out  = spike(o * (0.9*f*prev + 0.1*tanh(i*S)))
// ---------------------------------------------------------------------------

#define BATCH   4096
#define DIMD    1024
#define RESR    2048
#define KCOMB   3072   // D + R
#define MAXDIM  2560
#define NGATE   6144   // 3*R

#define TM      64     // workgroup M tile
#define TN      64     // workgroup N tile
#define TK      32     // K step (one bf16 WMMA)
#define LDT     40     // padded LDS row stride (halves): 80B, 16B aligned

typedef __attribute__((ext_vector_type(16))) __bf16 v16bf;
typedef __attribute__((ext_vector_type(8)))  float  v8f;

__device__ __forceinline__ unsigned short f32_to_bf16(float f) {
    uint32_t u = __float_as_uint(f);
    uint32_t r = u + 0x7FFFu + ((u >> 16) & 1u);   // round-to-nearest-even
    return (unsigned short)(r >> 16);
}

__device__ __forceinline__ float sigmoidf_fast(float x) {
    return 1.0f / (1.0f + __expf(-x));
}

// -------------------- conversion: f32 -> bf16, row-wise concat --------------
__global__ __launch_bounds__(256) void convert_concat_kernel(
    const float* __restrict__ a, int lda,
    const float* __restrict__ b, int ldb,
    int K0, int Ktot, int rows,
    unsigned short* __restrict__ dst)
{
    long long n   = (long long)rows * (long long)Ktot;
    long long idx = (long long)blockIdx.x * blockDim.x + threadIdx.x;
    long long stride = (long long)gridDim.x * blockDim.x;
    for (; idx < n; idx += stride) {
        int row = (int)(idx / Ktot);
        int k   = (int)(idx - (long long)row * Ktot);
        float v = (k < K0) ? a[(long long)row * lda + k]
                           : b[(long long)row * ldb + (k - K0)];
        dst[idx] = f32_to_bf16(v);
    }
}

// -------------------- zero the padding columns [R, MAX_DIM) -----------------
__global__ __launch_bounds__(256) void zero_pad_kernel(float* __restrict__ out)
{
    long long n   = (long long)BATCH * (MAXDIM - RESR);
    long long idx = (long long)blockIdx.x * blockDim.x + threadIdx.x;
    long long stride = (long long)gridDim.x * blockDim.x;
    for (; idx < n; idx += stride) {
        int row = (int)(idx / (MAXDIM - RESR));
        int col = RESR + (int)(idx - (long long)row * (MAXDIM - RESR));
        out[(long long)row * MAXDIM + col] = 0.0f;
    }
}

// -------------------- async global -> LDS tile fill (64 x 32 halves) --------
// One global_load_async_to_lds_b128 per thread: no VGPR round-trip, ASYNCcnt.
__device__ __forceinline__ void async_tile_64x32(
    unsigned short* ldsDst, const unsigned short* g, int ldg)
{
    const int t   = threadIdx.x;
    const int row = t >> 2;
    const int cg  = (t & 3) * 8;
    unsigned lds_off = (unsigned)(uintptr_t)(ldsDst + row * LDT + cg);
    const unsigned short* gp = g + (long long)row * ldg + cg;
    asm volatile("global_load_async_to_lds_b128 %0, %1, off"
                 :: "v"(lds_off), "v"(gp)
                 : "memory");
}

#define ASYNC_WAIT() asm volatile("s_wait_asynccnt 0x0" ::: "memory")

// -------------------- fragment loads (match ISA 7.12.2 layouts) -------------
// A 16x32 bf16 : lane L -> m = L%16, half h = L/16
//   VGPR0..3 : K = h*8 .. h*8+7   ; VGPR4..7 : K = 16+h*8 .. 16+h*8+7
__device__ __forceinline__ v16bf load_fragA(const unsigned short* rowp, int half)
{
    union { v16bf v; uint4 q[2]; } f;
    f.q[0] = *(const uint4*)(rowp + half * 8);
    f.q[1] = *(const uint4*)(rowp + 16 + half * 8);
    return f.v;
}
// B 32x16 bf16 : lane L -> n = L%16, half h = L/16
//   VGPR0..3 : K = h*8 .. h*8+7   ; VGPR4..7 : K = 16+h*8 .. 16+h*8+7
__device__ __forceinline__ v16bf load_fragB(const unsigned short* rowp, int half)
{
    union { v16bf v; uint4 q[2]; } f;
    f.q[0] = *(const uint4*)(rowp + half * 8);
    f.q[1] = *(const uint4*)(rowp + 16 + half * 8);
    return f.v;
}

#define WMMA_BF16(A, Bm, C) \
    __builtin_amdgcn_wmma_f32_16x16x32_bf16(false, (A), false, (Bm), (short)0, (C), false, false)

// -------------------- epilogue for one 16x16 tile ---------------------------
__device__ __forceinline__ void epilogue_tile(
    v8f accS, v8f accI, v8f accF, v8f accO,
    int row0, int col,
    const float* __restrict__ prev_state, float* __restrict__ out)
{
    #pragma unroll
    for (int r = 0; r < 8; ++r) {
        const int row = row0 + r;
        const float pv = prev_state[(long long)row * MAXDIM + col];
        const float s  = accS[r];
        const float ig = sigmoidf_fast(accI[r]);
        const float fg = sigmoidf_fast(accF[r]);
        const float og = sigmoidf_fast(accO[r]);
        float st = 0.9f * (fg * pv) + 0.1f * tanhf(ig * s);
        st *= og;
        st = (st > 0.5f) ? (st - 0.5f) : st;
        out[(long long)row * MAXDIM + col] = st;
    }
}

// -------------------- fused GEMM + epilogue ---------------------------------
// grid: (BATCH/TM, RESR/TN), block: 256 (8 waves, wave32)
// wave w: m_tile = w & 3 ; n_tiles = {(w>>2)*2, (w>>2)*2 + 1}  (4x4 tile grid)
// LDS: double-buffered 5-tile set, filled by async-to-LDS, consumed by WMMA.
__global__ __launch_bounds__(256) void reservoir_fused_gemm(
    const unsigned short* __restrict__ Acomb,   // [BATCH][KCOMB] bf16
    const unsigned short* __restrict__ Wcomb,   // [RESR][KCOMB]  bf16
    const unsigned short* __restrict__ Wgate,   // [NGATE][DIMD]  bf16
    const float* __restrict__ prev_state,       // [BATCH][MAXDIM] f32
    float* __restrict__ out)                    // [BATCH][MAXDIM] f32
{
    // tiles: 0=A, 1=Wcomb, 2..4=Wgate[i/f/o]
    __shared__ unsigned short sBuf[2][5][TM * LDT];   // 2 x 25 KB

    const int bm = blockIdx.x * TM;
    const int bn = blockIdx.y * TN;

    const int wave = threadIdx.x >> 5;
    const int lane = threadIdx.x & 31;
    const int m_t  = wave & 3;
    const int n_b  = (wave >> 2) * 2;
    const int lm   = lane & 15;
    const int half = lane >> 4;

    const v8f vzero = {0.f,0.f,0.f,0.f,0.f,0.f,0.f,0.f};
    v8f accS0 = vzero, accS1 = vzero;
    v8f accI0 = vzero, accI1 = vzero;
    v8f accF0 = vzero, accF1 = vzero;
    v8f accO0 = vzero, accO1 = vzero;

    const unsigned short* gA  = Acomb + (long long)bm * KCOMB;
    const unsigned short* gWc = Wcomb + (long long)bn * KCOMB;
    const unsigned short* gWg0 = Wgate + (long long)(0 * RESR + bn) * DIMD;
    const unsigned short* gWg1 = Wgate + (long long)(1 * RESR + bn) * DIMD;
    const unsigned short* gWg2 = Wgate + (long long)(2 * RESR + bn) * DIMD;

    const int aOff  = (m_t * 16 + lm) * LDT;
    const int bOff0 = ((n_b + 0) * 16 + lm) * LDT;
    const int bOff1 = ((n_b + 1) * 16 + lm) * LDT;

    auto stage_full = [&](int c, int kb) {
        async_tile_64x32(sBuf[c][0], gA  + kb, KCOMB);
        async_tile_64x32(sBuf[c][1], gWc + kb, KCOMB);
        async_tile_64x32(sBuf[c][2], gWg0 + kb, DIMD);
        async_tile_64x32(sBuf[c][3], gWg1 + kb, DIMD);
        async_tile_64x32(sBuf[c][4], gWg2 + kb, DIMD);
    };
    auto stage_small = [&](int c, int kb) {
        async_tile_64x32(sBuf[c][0], gA  + kb, KCOMB);
        async_tile_64x32(sBuf[c][1], gWc + kb, KCOMB);
    };
    auto compute_full = [&](int c) {
        v16bf a = load_fragA(&sBuf[c][0][aOff], half);
        v16bf b0 = load_fragB(&sBuf[c][1][bOff0], half);
        accS0 = WMMA_BF16(a, b0, accS0);
        v16bf b1 = load_fragB(&sBuf[c][1][bOff1], half);
        accS1 = WMMA_BF16(a, b1, accS1);
        v16bf bi0 = load_fragB(&sBuf[c][2][bOff0], half);
        accI0 = WMMA_BF16(a, bi0, accI0);
        v16bf bi1 = load_fragB(&sBuf[c][2][bOff1], half);
        accI1 = WMMA_BF16(a, bi1, accI1);
        v16bf bf0 = load_fragB(&sBuf[c][3][bOff0], half);
        accF0 = WMMA_BF16(a, bf0, accF0);
        v16bf bf1 = load_fragB(&sBuf[c][3][bOff1], half);
        accF1 = WMMA_BF16(a, bf1, accF1);
        v16bf bo0 = load_fragB(&sBuf[c][4][bOff0], half);
        accO0 = WMMA_BF16(a, bo0, accO0);
        v16bf bo1 = load_fragB(&sBuf[c][4][bOff1], half);
        accO1 = WMMA_BF16(a, bo1, accO1);
    };
    auto compute_small = [&](int c) {
        v16bf a = load_fragA(&sBuf[c][0][aOff], half);
        v16bf b0 = load_fragB(&sBuf[c][1][bOff0], half);
        accS0 = WMMA_BF16(a, b0, accS0);
        v16bf b1 = load_fragB(&sBuf[c][1][bOff1], half);
        accS1 = WMMA_BF16(a, b1, accS1);
    };

    // ---- software pipeline: fill(cur^1, kb+TK) overlaps compute(cur) ----
    int cur = 0;
    stage_full(0, 0);
    ASYNC_WAIT();
    __syncthreads();

    // phase 1: K = 0 .. DIMD-1 (all four GEMMs), stage next full set
    for (int kb = 0; kb < DIMD - TK; kb += TK) {
        stage_full(cur ^ 1, kb + TK);
        compute_full(cur);
        ASYNC_WAIT();
        __syncthreads();
        cur ^= 1;
    }
    // boundary: last full compute, stage first phase-2 (small) set
    stage_small(cur ^ 1, DIMD);
    compute_full(cur);
    ASYNC_WAIT();
    __syncthreads();
    cur ^= 1;

    // phase 2: K = DIMD .. KCOMB-1 (combined GEMM only)
    for (int kb = DIMD; kb < KCOMB - TK; kb += TK) {
        stage_small(cur ^ 1, kb + TK);
        compute_small(cur);
        ASYNC_WAIT();
        __syncthreads();
        cur ^= 1;
    }
    compute_small(cur);   // drain

    // ---- epilogue ----
    const int row0 = bm + m_t * 16 + half * 8;
    const int col0 = bn + (n_b + 0) * 16 + lm;
    const int col1 = bn + (n_b + 1) * 16 + lm;
    epilogue_tile(accS0, accI0, accF0, accO0, row0, col0, prev_state, out);
    epilogue_tile(accS1, accI1, accF1, accO1, row0, col1, prev_state, out);
}

// ---------------------------------------------------------------------------
extern "C" void kernel_launch(void* const* d_in, const int* in_sizes, int n_in,
                              void* d_out, int out_size, void* d_ws, size_t ws_size,
                              hipStream_t stream)
{
    (void)in_sizes; (void)n_in; (void)out_size; (void)ws_size;
    const float* inputs     = (const float*)d_in[0];   // [4096][1024]
    const float* prev_state = (const float*)d_in[1];   // [4096][2560]
    const float* W_in       = (const float*)d_in[2];   // [2048][1024]
    const float* W_res      = (const float*)d_in[3];   // [2048][2048]
    const float* W_gate     = (const float*)d_in[4];   // [6144][1024]
    float* out = (float*)d_out;                        // [4096][2560]

    // workspace layout (bf16)
    char* ws = (char*)d_ws;
    unsigned short* Acomb = (unsigned short*)ws;                                   // 4096*3072
    unsigned short* Wcomb = Acomb + (size_t)BATCH * KCOMB;                         // 2048*3072
    unsigned short* Wgbf  = Wcomb + (size_t)RESR  * KCOMB;                         // 6144*1024

    {
        long long n = (long long)BATCH * KCOMB;
        convert_concat_kernel<<<(int)((n + 255) / 256), 256, 0, stream>>>(
            inputs, DIMD, prev_state, MAXDIM, DIMD, KCOMB, BATCH, Acomb);
    }
    {
        long long n = (long long)RESR * KCOMB;
        convert_concat_kernel<<<(int)((n + 255) / 256), 256, 0, stream>>>(
            W_in, DIMD, W_res, RESR, DIMD, KCOMB, RESR, Wcomb);
    }
    {
        long long n = (long long)NGATE * DIMD;
        convert_concat_kernel<<<(int)((n + 255) / 256), 256, 0, stream>>>(
            W_gate, DIMD, W_gate, DIMD, DIMD, DIMD, NGATE, Wgbf);
    }
    {
        long long n = (long long)BATCH * (MAXDIM - RESR);
        zero_pad_kernel<<<(int)((n + 255) / 256), 256, 0, stream>>>(out);
    }

    dim3 grid(BATCH / TM, RESR / TN);   // 64 x 32
    reservoir_fused_gemm<<<grid, 256, 0, stream>>>(
        Acomb, Wcomb, Wgbf, prev_state, out);
}
